// Transformer_46067819217522
// MI455X (gfx1250) — matrix operand order repaired and checked by reading, not compiled, not run on hardware
//
#include <hip/hip_runtime.h>
#include <math.h>
#include <stdint.h>

// ---------------------------------------------------------------------------
// Graph-transformer forward for MI455X (gfx1250, wave32, WMMA 16x16x32 f16).
// f16 storage for all matrix operands (attn kept f16 between layers; fp32
// attn only for the final layer -> d_out), fused score+softmax, fused
// residual + tiny GEMM stages via LDS relayout, async global->LDS staging
// for the conv tiles, unconditional speculative prefetch on streamed operands.
// ---------------------------------------------------------------------------

#define B_    8
#define C_    3
#define IMG_  512
#define P_    16
#define DIM_  256
#define HEADS_ 8
#define DEPTH_ 4
#define DH_   32
#define G_    32
#define L_    1024
#define EC_   64
#define PD_   1024
#define TOKENS (B_ * L_)   // 8192

typedef __attribute__((ext_vector_type(16))) _Float16 v16h;
typedef __attribute__((ext_vector_type(8)))  _Float16 v8h;
typedef __attribute__((ext_vector_type(8)))  float    v8f;

// ---------------- WMMA helpers (CDNA5 wave32 layouts, ISA 7.12.2) ----------

__device__ __forceinline__ v8f wmma32(v16h a, v16h b, v8f c) {
  // D = A(16x32 f16) * B(32x16 f16) + C(16x16 f32)
  return __builtin_amdgcn_wmma_f32_16x16x32_f16(false, a, false, b, (short)0, c,
                                                false, false);
}

// A operand: 16x32 f16 tile from row-major A (lda elems). Lane l:
// row M=l%16, K = kb..kb+15 in elems 0..15,
// kb = (l>=16)*16.
__device__ __forceinline__ v16h load_a16x32(const _Float16* A, int lda, int m0, int k0) {
  int lane = threadIdx.x & 31;
  const _Float16* p = A + (size_t)(m0 + (lane & 15)) * lda + (k0 + ((lane >> 4) << 4));
  v8h lo = *(const v8h*)p;
  v8h hi = *(const v8h*)(p + 8);
  v16h r;
#pragma unroll
  for (int i = 0; i < 8; ++i) { r[i] = lo[i]; r[i + 8] = hi[i]; }
  return r;
}

// B operand: 32x16 tile read from Bt stored (N x K) row-major -> per lane one
// contiguous 32B chunk: lane l holds col n=l%16, K = (l/16)*16 + i.
__device__ __forceinline__ v16h load_b32x16(const _Float16* Bt, int ldb, int n0, int k0) {
  int lane = threadIdx.x & 31;
  const _Float16* p = Bt + (size_t)(n0 + (lane & 15)) * ldb + (k0 + ((lane >> 4) << 4));
  return *(const v16h*)p;
}

__device__ __forceinline__ v8f zero8() {
  v8f z = {0.f, 0.f, 0.f, 0.f, 0.f, 0.f, 0.f, 0.f};
  return z;
}

__device__ __forceinline__ float gelu_exact(float x) {
  return 0.5f * x * (1.f + erff(x * 0.70710678118654752f));
}

// ---------------- CDNA5 async global->LDS copy (ASYNCcnt path) -------------
// Each active lane copies 16 contiguous bytes global -> LDS, no VGPR bounce.
__device__ __forceinline__ void async_copy16(const void* gptr, void* lptr) {
#if defined(__gfx1250__)
  uint64_t g = (uint64_t)(uintptr_t)gptr;
  uint32_t l = (uint32_t)(uintptr_t)lptr;   // LDS byte offset in low bits
  asm volatile("global_load_async_to_lds_b128 %0, %1, off"
               :: "v"(l), "v"(g) : "memory");
#else
  (void)gptr; (void)lptr;
#endif
}

__device__ __forceinline__ void async_wait0() {
#if defined(__gfx1250__)
  asm volatile("s_wait_asynccnt 0" ::: "memory");
#endif
}

// ---------------- small utility kernels ------------------------------------

__global__ void k_zero(float* p, int n) {
  int i = blockIdx.x * 256 + threadIdx.x;
  if (i < n) p[i] = 0.f;
}

// out (N x K) f16 = transpose of in (K x N) f32
__global__ void k_transpose_h(const float* in, _Float16* out, int K, int N) {
  int idx = blockIdx.x * 256 + threadIdx.x;
  if (idx >= K * N) return;
  int n = idx / K, k = idx - n * K;
  out[idx] = (_Float16)in[k * N + n];
}

__global__ void k_convert_h(const float* in, _Float16* out, int n) {
  int i = blockIdx.x * 256 + threadIdx.x;
  if (i < n) out[i] = (_Float16)in[i];
}

// ---------------- patch conv + BN stats ------------------------------------

__device__ __forceinline__ float conv_one(const float* sw, const float* sin,
                                          int ch, int py, int px, float bias) {
  float acc = bias;
#pragma unroll
  for (int c = 0; c < 3; ++c) {
#pragma unroll
    for (int ky = 0; ky < 3; ++ky) {
      int iy = 2 * py - 1 + ky;
      if (iy < 0 || iy > 15) continue;
#pragma unroll
      for (int kx = 0; kx < 3; ++kx) {
        int ix = 2 * px - 1 + kx;
        if (ix < 0 || ix > 15) continue;
        acc += sw[ch * 27 + c * 9 + ky * 3 + kx] * sin[c * 256 + iy * 16 + ix];
      }
    }
  }
  return acc;
}

// Cooperative async staging of conv weights (1728 f32) + one input patch
// (768 f32) into LDS via GLOBAL_LOAD_ASYNC_TO_LDS_B128.
__device__ __forceinline__ void stage_conv_tiles(const float* img, const float* conv_w,
                                                 float* sw, float* sin,
                                                 int b, int gy, int gx) {
  int t = threadIdx.x;
  for (int i = t; i < 432; i += 256)              // 432 * 16B = 6912B weights
    async_copy16(conv_w + i * 4, sw + i * 4);
  if (t < 192) {                                  // 48 rows * 64B patch input
    int i = t;
    int row = i >> 2, part = i & 3;               // row = c*16 + py
    int c = row >> 4, py = row & 15;
    async_copy16(&img[((size_t)(b * 3 + c) * 512 + gy * 16 + py) * 512 + gx * 16 + part * 4],
                 sin + row * 16 + part * 4);
  }
  async_wait0();
}

__global__ void k_conv_stats(const float* img, const float* conv_w, const float* conv_b,
                             float* stats /* [64]=sum [64..127]=sumsq */) {
  __shared__ __align__(16) float sw[64 * 27];
  __shared__ __align__(16) float sin[3 * 256];
  __shared__ float ssum[64], ssq[64];
  int t = threadIdx.x;
  int patch = blockIdx.x;            // b*1024 + gy*32 + gx
  int b = patch >> 10;
  int l = patch & 1023;
  int gy = l >> 5, gx = l & 31;
  stage_conv_tiles(img, conv_w, sw, sin, b, gy, gx);
  if (t < 64) { ssum[t] = 0.f; ssq[t] = 0.f; }
  __syncthreads();
  int ch = t >> 2;
  float bias = conv_b[ch];
  float s = 0.f, sq = 0.f;
#pragma unroll
  for (int i = 0; i < 16; ++i) {
    int p = (t & 3) * 16 + i;
    float v = conv_one(sw, sin, ch, p >> 3, p & 7, bias);
    s += v; sq += v * v;
  }
  atomicAdd(&ssum[ch], s);
  atomicAdd(&ssq[ch], sq);
  __syncthreads();
  if (t < 64) { atomicAdd(&stats[t], ssum[t]); atomicAdd(&stats[64 + t], ssq[t]); }
}

__global__ void k_bn_finalize(float* stats) {
  int c = threadIdx.x;
  if (c >= 64) return;
  const float N = (float)TOKENS * 64.f;   // 8192 patches * 8*8 pixels
  float mean = stats[c] / N;
  float var = stats[64 + c] / N - mean * mean;
  stats[c] = mean;
  stats[64 + c] = rsqrtf(var + 1e-5f);
}

// conv (recomputed) + BN + gelu + maxpool -> x0h f16 (TOKENS x PD)
__global__ void k_patch_embed(const float* img, const float* conv_w, const float* conv_b,
                              const float* stats, const float* bn_g, const float* bn_b,
                              _Float16* x0h) {
  __shared__ __align__(16) float sw[64 * 27];
  __shared__ __align__(16) float sin[3 * 256];
  __shared__ float sconv[64 * 64];   // 64ch x 8x8 post-gelu
  int t = threadIdx.x;
  int patch = blockIdx.x;
  int b = patch >> 10;
  int l = patch & 1023;
  int gy = l >> 5, gx = l & 31;
  stage_conv_tiles(img, conv_w, sw, sin, b, gy, gx);
  __syncthreads();
  int ch = t >> 2;
  float bias = conv_b[ch];
  float mean = stats[ch], rstd = stats[64 + ch];
  float gsc = bn_g[ch] * rstd, bof = bn_b[ch];
#pragma unroll
  for (int i = 0; i < 16; ++i) {
    int p = (t & 3) * 16 + i;
    float v = conv_one(sw, sin, ch, p >> 3, p & 7, bias);
    v = (v - mean) * gsc + bof;
    sconv[ch * 64 + p] = gelu_exact(v);
  }
  __syncthreads();
  // maxpool 3x3 s2 p1 on 8x8 -> 4x4 ; flatten order (ch, y, x)
  for (int i = t; i < 1024; i += 256) {
    int c2 = i >> 4, p = i & 15, oy = p >> 2, ox = p & 3;
    float m = -INFINITY;
#pragma unroll
    for (int ky = 0; ky < 3; ++ky) {
      int y = 2 * oy - 1 + ky;
      if (y < 0 || y > 7) continue;
#pragma unroll
      for (int kx = 0; kx < 3; ++kx) {
        int x = 2 * ox - 1 + kx;
        if (x < 0 || x > 7) continue;
        m = fmaxf(m, sconv[c2 * 64 + y * 8 + x]);
      }
    }
    x0h[(size_t)patch * PD_ + i] = (_Float16)m;
  }
}

// ---------------- generic WMMA GEMM: out = A@Bt^T (+bias)(+res) ------------
// A: M x K f16 row-major (lda=K). Bt: N x K f16. out fp32 M x N.
// grid (M/64, N/64), block 128 (4 waves); wave w -> rows w*16..w*16+15.
__global__ void k_gemm(const _Float16* __restrict__ A, const _Float16* __restrict__ Bt,
                       const float* bias, const float* res, float* out, int N, int K) {
  int wave = threadIdx.x >> 5, lane = threadIdx.x & 31;
  int m0 = blockIdx.x * 64 + wave * 16;
  int n0 = blockIdx.y * 64;
  const _Float16* arow = A + (size_t)(m0 + (lane & 15)) * K;
  v8f acc[4];
#pragma unroll
  for (int t2 = 0; t2 < 4; ++t2) acc[t2] = zero8();
  for (int k0 = 0; k0 < K; k0 += 32) {
    // Unconditional speculative prefetch 512B ahead (DEV scope: OOB at the
    // stream tail is silently dropped; no counters touched, no branch).
    __builtin_prefetch(arow + k0 + 256, 0, 1);
    v16h a = load_a16x32(A, K, m0, k0);
#pragma unroll
    for (int t2 = 0; t2 < 4; ++t2) {
      v16h b = load_b32x16(Bt, K, n0 + t2 * 16, k0);
      acc[t2] = wmma32(a, b, acc[t2]);
    }
  }
  int colb = n0 + (lane & 15);
  int rh = (lane >> 4) << 3;
#pragma unroll
  for (int t2 = 0; t2 < 4; ++t2) {
    int col = colb + t2 * 16;
    float bv = bias ? bias[col] : 0.f;
#pragma unroll
    for (int r = 0; r < 8; ++r) {
      int row = m0 + rh + r;
      float v = acc[t2][r] + bv;
      if (res) v += res[(size_t)row * N + col];
      out[(size_t)row * N + col] = v;
    }
  }
}

// ---------------- layernorm -> xn fp32, xhT f16 (b,h,d,t) -------------------
__global__ void k_layernorm(const float* x, const float* g, const float* bb,
                            float* xn, _Float16* xhT) {
  __shared__ float red[256];
  int t = threadIdx.x, tok = blockIdx.x;
  float v = x[(size_t)tok * 256 + t];
  red[t] = v;
  __syncthreads();
  for (int s = 128; s > 0; s >>= 1) { if (t < s) red[t] += red[t + s]; __syncthreads(); }
  float mean = red[0] * (1.f / 256.f);
  __syncthreads();
  float d = v - mean;
  red[t] = d * d;
  __syncthreads();
  for (int s = 128; s > 0; s >>= 1) { if (t < s) red[t] += red[t + s]; __syncthreads(); }
  float rstd = rsqrtf(red[0] * (1.f / 256.f) + 1e-5f);
  float o = d * rstd * g[t] + bb[t];
  xn[(size_t)tok * 256 + t] = o;
  int b = tok >> 10, tt = tok & 1023;
  int h = t >> 5, dd = t & 31;
  xhT[(((size_t)(b * 8 + h)) * 32 + dd) * 1024 + tt] = (_Float16)o;
}

// ---------------- m = A@xh + xh ; qk = relu(m@qk_w + b) -> q,k f16 ----------
// grid (L/64, B*H), block 128.
__global__ void k_graph_qk(const _Float16* __restrict__ Ah, const _Float16* __restrict__ xhT,
                           const float* __restrict__ xn,
                           const _Float16* __restrict__ qkT, const float* __restrict__ qkb,
                           _Float16* qh, _Float16* kh) {
  __shared__ __align__(16) _Float16 smt[4][16 * 32];
  int wave = threadIdx.x >> 5, lane = threadIdx.x & 31;
  int bh = blockIdx.y, b = bh >> 3, h = bh & 7;
  int s0 = blockIdx.x * 64 + wave * 16;
  const _Float16* BT = xhT + (size_t)bh * 32 * 1024;
  const _Float16* arow = Ah + (size_t)(s0 + (lane & 15)) * 1024;
  v8f acc[2] = {zero8(), zero8()};
  for (int k0 = 0; k0 < 1024; k0 += 32) {
    __builtin_prefetch(arow + k0 + 256, 0, 1);
    v16h a = load_a16x32(Ah, 1024, s0, k0);
#pragma unroll
    for (int t2 = 0; t2 < 2; ++t2)
      acc[t2] = wmma32(a, load_b32x16(BT, 1024, t2 * 16, k0), acc[t2]);
  }
  int col = lane & 15, rh = (lane >> 4) << 3;
  _Float16* S = smt[wave];
#pragma unroll
  for (int t2 = 0; t2 < 2; ++t2)
#pragma unroll
    for (int r = 0; r < 8; ++r) {
      int srow = s0 + rh + r;
      float mval = acc[t2][r] + xn[(size_t)((b << 10) + srow) * 256 + (h << 5) + t2 * 16 + col];
      S[(rh + r) * 32 + t2 * 16 + col] = (_Float16)mval;
    }
  __syncthreads();
  v16h am = load_a16x32(S, 32, 0, 0);   // LDS -> A-layout (ds loads)
  v8f q4[4];
#pragma unroll
  for (int t2 = 0; t2 < 4; ++t2)
    q4[t2] = wmma32(am, load_b32x16(qkT, 32, t2 * 16, 0), zero8());
#pragma unroll
  for (int t2 = 0; t2 < 4; ++t2) {
    int n = t2 * 16 + col;
    float bv = qkb[n];
#pragma unroll
    for (int r = 0; r < 8; ++r) {
      int srow = s0 + rh + r;
      float v = fmaxf(q4[t2][r] + bv, 0.f);
      if (n < 32) qh[(size_t)bh * 32768 + srow * 32 + n] = (_Float16)v;
      else        kh[(size_t)bh * 32768 + srow * 32 + (n - 32)] = (_Float16)v;
    }
  }
}

// ---------------- fused score GEMM + softmax -> attn f16 (+fp32 out) -------
// grid (L/16, B*H), block 256 (8 waves, each owns 128 cols of one 16-row strip)
__global__ void k_score_softmax(const _Float16* __restrict__ qh, const _Float16* __restrict__ kh,
                                _Float16* attn_h, float* attn_out) {
  __shared__ float smax[8 * 16];
  __shared__ float ssum[8 * 16];
  int wave = threadIdx.x >> 5, lane = threadIdx.x & 31;
  int bh = blockIdx.y;
  int s0 = blockIdx.x << 4;
  const _Float16* Q = qh + (size_t)bh * 32768;
  const _Float16* Kt = kh + (size_t)bh * 32768;
  v16h a = load_a16x32(Q, 32, s0, 0);
  v8f acc[8];
#pragma unroll
  for (int t2 = 0; t2 < 8; ++t2)
    acc[t2] = wmma32(a, load_b32x16(Kt, 32, wave * 128 + t2 * 16, 0), zero8());
  const float sc = 1.f / 32.f;   // DIM^-0.5 / 2
  int half = lane >> 4;
  float pm[8];
#pragma unroll
  for (int r = 0; r < 8; ++r) {
    float m = acc[0][r] * sc;
    acc[0][r] = m;
#pragma unroll
    for (int t2 = 1; t2 < 8; ++t2) { acc[t2][r] *= sc; m = fmaxf(m, acc[t2][r]); }
#pragma unroll
    for (int off = 8; off > 0; off >>= 1) m = fmaxf(m, __shfl_xor(m, off, 32));
    pm[r] = m;
  }
  if ((lane & 15) == 0)
#pragma unroll
    for (int r = 0; r < 8; ++r) smax[wave * 16 + half * 8 + r] = pm[r];
  __syncthreads();
  float ps[8];
#pragma unroll
  for (int r = 0; r < 8; ++r) {
    int row = half * 8 + r;
    float m = -INFINITY;
#pragma unroll
    for (int wv = 0; wv < 8; ++wv) m = fmaxf(m, smax[wv * 16 + row]);
    float s = 0.f;
#pragma unroll
    for (int t2 = 0; t2 < 8; ++t2) {
      float e = __expf(acc[t2][r] - m);
      acc[t2][r] = e;
      s += e;
    }
#pragma unroll
    for (int off = 8; off > 0; off >>= 1) s += __shfl_xor(s, off, 32);
    ps[r] = s;
  }
  if ((lane & 15) == 0)
#pragma unroll
    for (int r = 0; r < 8; ++r) ssum[wave * 16 + half * 8 + r] = ps[r];
  __syncthreads();
  int colb = wave * 128 + (lane & 15);
#pragma unroll
  for (int r = 0; r < 8; ++r) {
    int row = half * 8 + r;
    float tot = 0.f;
#pragma unroll
    for (int wv = 0; wv < 8; ++wv) tot += ssum[wv * 16 + row];
    float inv = 1.f / tot;
    size_t base = (size_t)bh * 1048576 + (size_t)(s0 + row) * 1024;
#pragma unroll
    for (int t2 = 0; t2 < 8; ++t2) {
      float v = acc[t2][r] * inv;
      attn_h[base + colb + t2 * 16] = (_Float16)v;
      if (attn_out) attn_out[base + colb + t2 * 16] = v;
    }
  }
}

// ---------------- mv = attn@xh + xh ; v = gelu(relu(mv@v_w+b)) -> vbuf -----
__global__ void k_mv_v(const _Float16* __restrict__ attn_h, const _Float16* __restrict__ xhT,
                       const float* __restrict__ xn,
                       const _Float16* __restrict__ vT, const float* __restrict__ vb,
                       _Float16* vbuf) {
  __shared__ __align__(16) _Float16 smt[4][16 * 32];
  int wave = threadIdx.x >> 5, lane = threadIdx.x & 31;
  int bh = blockIdx.y, b = bh >> 3, h = bh & 7;
  int s0 = blockIdx.x * 64 + wave * 16;
  const _Float16* A = attn_h + (size_t)bh * 1048576;
  const _Float16* BT = xhT + (size_t)bh * 32 * 1024;
  const _Float16* arow = A + (size_t)(s0 + (lane & 15)) * 1024;
  v8f acc[2] = {zero8(), zero8()};
  for (int k0 = 0; k0 < 1024; k0 += 32) {
    __builtin_prefetch(arow + k0 + 256, 0, 1);   // stream the attn operand
    v16h a = load_a16x32(A, 1024, s0, k0);
#pragma unroll
    for (int t2 = 0; t2 < 2; ++t2)
      acc[t2] = wmma32(a, load_b32x16(BT, 1024, t2 * 16, k0), acc[t2]);
  }
  int col = lane & 15, rh = (lane >> 4) << 3;
  _Float16* S = smt[wave];
#pragma unroll
  for (int t2 = 0; t2 < 2; ++t2)
#pragma unroll
    for (int r = 0; r < 8; ++r) {
      int srow = s0 + rh + r;
      float mv = acc[t2][r] + xn[(size_t)((b << 10) + srow) * 256 + (h << 5) + t2 * 16 + col];
      S[(rh + r) * 32 + t2 * 16 + col] = (_Float16)mv;
    }
  __syncthreads();
  v16h am = load_a16x32(S, 32, 0, 0);
  v8f v2[2];
#pragma unroll
  for (int t2 = 0; t2 < 2; ++t2)
    v2[t2] = wmma32(am, load_b32x16(vT, 32, t2 * 16, 0), zero8());
#pragma unroll
  for (int t2 = 0; t2 < 2; ++t2) {
    int n = t2 * 16 + col;
    float bv = vb[n];
#pragma unroll
    for (int r = 0; r < 8; ++r) {
      int srow = s0 + rh + r;
      float v = gelu_exact(fmaxf(v2[t2][r] + bv, 0.f));
      vbuf[(size_t)((b << 10) + srow) * 256 + (h << 5) + n] = (_Float16)v;
    }
  }
}

// ---------------------------------------------------------------------------

extern "C" void kernel_launch(void* const* d_in, const int* in_sizes, int n_in,
                              void* d_out, int out_size, void* d_ws, size_t ws_size,
                              hipStream_t stream) {
  const float* img    = (const float*)d_in[0];
  const float* adj    = (const float*)d_in[1];
  const float* conv_w = (const float*)d_in[3];
  const float* conv_b = (const float*)d_in[4];
  const float* bn_g   = (const float*)d_in[5];
  const float* bn_b   = (const float*)d_in[6];
  const float* emb_w  = (const float*)d_in[7];
  const float* emb_b  = (const float*)d_in[8];
  const float* ln_g   = (const float*)d_in[9];
  const float* ln_b   = (const float*)d_in[10];
  const float* qk_w   = (const float*)d_in[11];
  const float* qk_b   = (const float*)d_in[12];
  const float* v_w    = (const float*)d_in[13];
  const float* v_b    = (const float*)d_in[14];
  const float* proj_w = (const float*)d_in[15];
  float* out = (float*)d_out;

  char* ws = (char*)d_ws;
  size_t off = 0;
  auto alloc = [&](size_t bytes) -> void* {
    void* p = ws + off;
    off = (off + bytes + 255) & ~(size_t)255;
    return p;
  };

  float*     stats = (float*)alloc(128 * sizeof(float));
  _Float16*  x0h   = (_Float16*)alloc((size_t)TOKENS * PD_ * 2);
  _Float16*  Ah    = (_Float16*)alloc((size_t)L_ * L_ * 2);
  _Float16*  embT  = (_Float16*)alloc((size_t)DIM_ * PD_ * 2);
  _Float16*  qkT   = (_Float16*)alloc((size_t)DEPTH_ * 64 * 32 * 2);
  _Float16*  vT    = (_Float16*)alloc((size_t)DEPTH_ * 32 * 32 * 2);
  _Float16*  projT = (_Float16*)alloc((size_t)DEPTH_ * DIM_ * DIM_ * 2);
  float*     x     = (float*)alloc((size_t)TOKENS * DIM_ * 4);
  float*     xn    = (float*)alloc((size_t)TOKENS * DIM_ * 4);
  _Float16*  xhT   = (_Float16*)alloc((size_t)B_ * HEADS_ * DH_ * L_ * 2);
  _Float16*  qh    = (_Float16*)alloc((size_t)B_ * HEADS_ * L_ * DH_ * 2);
  _Float16*  kh    = (_Float16*)alloc((size_t)B_ * HEADS_ * L_ * DH_ * 2);
  _Float16*  attnH = (_Float16*)alloc((size_t)B_ * HEADS_ * L_ * L_ * 2);
  _Float16*  vbuf  = (_Float16*)alloc((size_t)TOKENS * DIM_ * 2);

  // --- weight prep -----------------------------------------------------
  k_zero<<<1, 256, 0, stream>>>(stats, 128);
  k_convert_h<<<(L_ * L_ + 255) / 256, 256, 0, stream>>>(adj, Ah, L_ * L_);
  k_transpose_h<<<(PD_ * DIM_ + 255) / 256, 256, 0, stream>>>(emb_w, embT, PD_, DIM_);
  for (int i = 0; i < DEPTH_; ++i) {
    k_transpose_h<<<(32 * 64 + 255) / 256, 256, 0, stream>>>(qk_w + i * 32 * 64,
                                                             qkT + i * 64 * 32, 32, 64);
    k_transpose_h<<<(32 * 32 + 255) / 256, 256, 0, stream>>>(v_w + i * 32 * 32,
                                                             vT + i * 32 * 32, 32, 32);
    k_transpose_h<<<(DIM_ * DIM_ + 255) / 256, 256, 0, stream>>>(proj_w + i * DIM_ * DIM_,
                                                                 projT + i * DIM_ * DIM_,
                                                                 DIM_, DIM_);
  }

  // --- conv / BN / pool / embed ----------------------------------------
  k_conv_stats<<<TOKENS, 256, 0, stream>>>(img, conv_w, conv_b, stats);
  k_bn_finalize<<<1, 64, 0, stream>>>(stats);
  k_patch_embed<<<TOKENS, 256, 0, stream>>>(img, conv_w, conv_b, stats, bn_g, bn_b, x0h);
  {
    dim3 g(TOKENS / 64, DIM_ / 64);
    k_gemm<<<g, 128, 0, stream>>>(x0h, embT, emb_b, nullptr, x, DIM_, PD_);
  }

  // --- transformer layers ----------------------------------------------
  for (int i = 0; i < DEPTH_; ++i) {
    k_layernorm<<<TOKENS, 256, 0, stream>>>(x, ln_g + i * DIM_, ln_b + i * DIM_, xn, xhT);
    {
      dim3 g(L_ / 64, B_ * HEADS_);
      k_graph_qk<<<g, 128, 0, stream>>>(Ah, xhT, xn, qkT + i * 64 * 32, qk_b + i * 64,
                                        qh, kh);
    }
    {
      dim3 g(L_ / 16, B_ * HEADS_);
      k_score_softmax<<<g, 256, 0, stream>>>(qh, kh, attnH,
                                             (i == DEPTH_ - 1) ? out : nullptr);
    }
    if (i == DEPTH_ - 1) break;   // only attn of last layer is the output
    {
      dim3 g(L_ / 64, B_ * HEADS_);
      k_mv_v<<<g, 128, 0, stream>>>(attnH, xhT, xn, vT + i * 32 * 32, v_b + i * 32, vbuf);
    }
    {
      dim3 g(TOKENS / 64, DIM_ / 64);
      k_gemm<<<g, 128, 0, stream>>>(vbuf, projT + i * DIM_ * DIM_, nullptr, xn, x,
                                    DIM_, DIM_);
    }
  }
  (void)in_sizes; (void)n_in; (void)out_size; (void)ws_size;
}
